// QuantumEnhancedStabilitySystem_84138409329310
// MI455X (gfx1250) — matrix operand order repaired and checked by reading, not compiled, not run on hardware
//
#include <hip/hip_runtime.h>
#include <cstdint>
#include <cstddef>

typedef __attribute__((ext_vector_type(16))) __bf16 v16bf;
typedef __attribute__((ext_vector_type(8)))  float  v8f;
typedef __attribute__((ext_vector_type(2)))  float  v2f;
typedef int v4i_vs __attribute__((vector_size(16)));      /* matches async builtin param */
typedef uint32_t v4u_vs __attribute__((vector_size(16))); /* TDM D# group0 (4 SGPRs) */
typedef uint32_t v8u_vs __attribute__((vector_size(32))); /* TDM D# group1 (8 SGPRs) */

#if defined(__has_builtin)
#if __has_builtin(__builtin_amdgcn_wmma_f32_16x16x4_f32)
#define HAS_WMMA_F32 1
#endif
#if __has_builtin(__builtin_amdgcn_global_load_async_to_lds_b128) && \
    __has_builtin(__builtin_amdgcn_global_store_async_from_lds_b128) && \
    __has_builtin(__builtin_amdgcn_s_wait_asynccnt)
#define HAS_ASYNC_LDS 1
#endif
#endif

#if defined(__AMDGCN__)
#define HAS_TDM 1          /* inline-asm tensor_load_to_lds; device pass only */
#endif

/* pointer casts for the async-to-LDS builtins (b128: v4i pointee) */
#define GPTR128(p) ((__attribute__((address_space(1))) v4i_vs*)(p))
#define LPTR128(p) ((__attribute__((address_space(3))) v4i_vs*)(p))

#define QN        256
#define BATCH     128
#define KTOT      65536      /* magnitude half of the 131072 features */
#define W1STRIDE  131072
#define N1        128
#define NS_ITERS  30
#define KSEGS     64         /* split-K segments -> 8*64 = 512 workgroups */

/* workspace layout (byte offsets) */
#define WS_H1      0            /* 128*128 f32  = 65536 B  */
#define WS_NEEDED  65536        /* 128 int                 */
#define WS_IDX     66048        /* 128 int                 */
#define WS_X0      131072       /* 3*256*256 f32 = 786432 B */
#define WS_X1      917504
#define WS_T       1703936      /* end = 2490368 B */

/* ---------------- h1 = bias init (split-K accumulates on top) ------------- */
__global__ void init_h1_kernel(const float* __restrict__ b1, float* __restrict__ h1) {
    int i = blockIdx.x * blockDim.x + threadIdx.x;
    if (i < BATCH * N1) h1[i] = b1[i & (N1 - 1)];
}

/* --------- GEMM1: h1 += |flat| @ W1[:, :65536].T  (bf16 WMMA, split-K) ---- */
/* grid (8 m-tiles, KSEGS k-segments), 256 threads = 8 waves; wave w owns n-tile w */
__global__ __launch_bounds__(256) void gemm1_kernel(const float* __restrict__ flat,
                                                    const float* __restrict__ W1,
                                                    float* __restrict__ h1,
                                                    int kseg_len) {
    /* raw f32 A tile; 260-float row stride (TDM pads 4 DWORDs per 256) */
    __shared__ float Asf[16][260];
    const int mtile  = blockIdx.x;
    const int koff0  = blockIdx.y * kseg_len;
    const int tid    = threadIdx.x;
    const int wave   = tid >> 5;
    const int lane   = tid & 31;
    const int l15    = lane & 15;
    const int lhi    = lane >> 4;
    const int nglob  = wave * 16 + l15;

    v8f acc = {};
    for (int kc = 0; kc < kseg_len; kc += 256) {
        const int koff = koff0 + kc;
        /* ---- stage A chunk (16 rows x 256 f32) into LDS ---- */
#ifdef HAS_TDM
        if (wave == 0) {
            /* D# group0: count=1 | lds_addr | global_addr[56:0] | type=2 */
            const uint64_t ga = (uint64_t)(uintptr_t)(flat + (size_t)(mtile * 16) * KTOT + koff);
            const uint32_t lds_base =
                (uint32_t)(size_t)(__attribute__((address_space(3))) float*)&Asf[0][0];
            v4u_vs g0 = { 1u,                                   /* count=1, user mode   */
                          lds_base,                             /* lds_addr (bytes)     */
                          (uint32_t)ga,                         /* global_addr[31:0]    */
                          (uint32_t)((ga >> 32) & 0x1FFFFFFu)   /* global_addr[56:32]   */
                          | 0x80000000u };                      /* type=2 ("image")     */
            /* D# group1: data_size=4B; pad_enable, interval=256dw, amount=4dw;
               tensor_dim0=65536, tensor_dim1=128, tile 256x16, dim0_stride=65536 */
            v8u_vs g1 = { 0x07D20000u,
                          0x00000000u,                          /* tensor_dim0[15:0]=0  */
                          (65536u >> 16) | (128u << 16),        /* dim0 hi | dim1 lo    */
                          (256u << 16),                         /* tile_dim0=256        */
                          16u,                                  /* tile_dim1=16         */
                          65536u,                               /* dim0_stride[31:0]    */
                          0u, 0u };
            asm volatile("tensor_load_to_lds %0, %1" :: "s"(g0), "s"(g1) : "memory");
        }
        __builtin_amdgcn_s_wait_tensorcnt(0);
#else
        {
            const int r  = tid >> 4;
            const int cb = (tid & 15) * 16;
            const float* src = flat + (size_t)(mtile * 16 + r) * KTOT + koff + cb;
#ifdef HAS_ASYNC_LDS
            #pragma unroll
            for (int j = 0; j < 4; ++j)
                __builtin_amdgcn_global_load_async_to_lds_b128(
                    GPTR128(src + j * 4), LPTR128(&Asf[r][cb + j * 4]), 0, 0);
            __builtin_amdgcn_s_wait_asynccnt(0);
#else
            #pragma unroll
            for (int j = 0; j < 16; ++j) Asf[r][cb + j] = src[j];
#endif
        }
#endif
        __syncthreads();
        /* prefetch next chunk of this wave's W1 rows into cache */
        __builtin_prefetch(W1 + (size_t)nglob * W1STRIDE + koff + 256, 0, 1);
        #pragma unroll
        for (int ks = 0; ks < 8; ++ks) {
            /* A fragment: 16-bit A 16x32 layout (ISA 7.12.2):
               lane half selects K+8 offset; elems 0..7 -> K=b..b+7, 8..15 -> K=b+16..b+23 */
            const int kbA = ks * 32 + lhi * 8;
            v16bf a;
            #pragma unroll
            for (int j = 0; j < 8; ++j) a[j]     = (__bf16)__builtin_fabsf(Asf[l15][kbA + j]);
            #pragma unroll
            for (int j = 0; j < 8; ++j) a[8 + j] = (__bf16)__builtin_fabsf(Asf[l15][kbA + 16 + j]);
            /* B fragment: same K order as A (elems 0..7 -> K=b..b+7, 8..15 -> K=b+16..b+23); N = lane&15 */
            const int kbB = koff + ks * 32 + lhi * 8;
            const float* wp = W1 + (size_t)nglob * W1STRIDE + kbB;
            v16bf b;
            #pragma unroll
            for (int j = 0; j < 8; ++j) b[j] = (__bf16)wp[j];
            #pragma unroll
            for (int j = 0; j < 8; ++j) b[8 + j] = (__bf16)wp[16 + j];
            acc = __builtin_amdgcn_wmma_f32_16x16x32_bf16(false, a, false, b,
                                                          (short)0, acc, false, false);
        }
        __syncthreads();
    }
    /* C layout: VGPR r -> M = r (lanes 0-15) / 8+r (lanes 16-31), N = lane&15 */
    #pragma unroll
    for (int r = 0; r < 8; ++r) {
        const int m = mtile * 16 + r + lhi * 8;
        atomicAdd(&h1[m * N1 + nglob], acc[r]);
    }
}

/* -------- tiny MLP: h2=relu(relu(h1)@W2.T+b2); syn=h2@W3.T+b3; flags ------ */
__global__ __launch_bounds__(64) void mlp_kernel(const float* __restrict__ h1,
                                                 const float* __restrict__ W2,
                                                 const float* __restrict__ b2,
                                                 const float* __restrict__ W3,
                                                 const float* __restrict__ b3,
                                                 int* __restrict__ needed,
                                                 int* __restrict__ idxout) {
    __shared__ float h2s[64];
    __shared__ float syn[8];
    const int b = blockIdx.x, t = threadIdx.x;
    float acc = b2[t];
    for (int k = 0; k < 128; ++k) {
        float h = h1[b * 128 + k];
        h = h > 0.f ? h : 0.f;                 /* relu on layer-1 output */
        acc += h * W2[t * 128 + k];
    }
    h2s[t] = acc > 0.f ? acc : 0.f;
    __syncthreads();
    if (t < 8) {
        float s = b3[t];
        for (int k = 0; k < 64; ++k) s += h2s[k] * W3[t * 64 + k];
        syn[t] = s;
    }
    __syncthreads();
    if (t == 0) {
        float best = -1.f; int bi = 0, nd = 0;
        for (int j = 0; j < 8; ++j) {
            float a = fabsf(syn[j]);
            if (a > 0.0001f) nd = 1;
            if (a > best) { best = a; bi = j; }   /* first max, like jnp.argmax */
        }
        needed[b] = nd;
        idxout[b] = bi % 3;
    }
}

/* --------------- X0 = C / ||C||_F for each of the 3 ops ------------------- */
__global__ __launch_bounds__(256) void polar_init_kernel(const float* __restrict__ ops,
                                                         float* __restrict__ X0) {
    __shared__ float red[256];
    const int m = blockIdx.x, t = threadIdx.x;
    const float* C = ops + (size_t)m * QN * QN;
    float s = 0.f;
    for (int i = t; i < QN * QN; i += 256) { float v = C[i]; s += v * v; }
    red[t] = s;
    __syncthreads();
    for (int off = 128; off > 0; off >>= 1) {
        if (t < off) red[t] += red[t + off];
        __syncthreads();
    }
    const float inv = rsqrtf(red[0] + 1e-30f);
    float* X = X0 + (size_t)m * QN * QN;
    for (int i = t; i < QN * QN; i += 256) X[i] = C[i] * inv;
}

/* ---------------- Newton–Schulz: T = X^T X  (f32 WMMA) -------------------- */
__global__ __launch_bounds__(32) void ns_t_kernel(const float* __restrict__ Xall,
                                                  float* __restrict__ Tall) {
    const float* X = Xall + (size_t)blockIdx.z * QN * QN;
    float* T = Tall + (size_t)blockIdx.z * QN * QN;
    const int mtile = blockIdx.y, ntile = blockIdx.x;
    const int lane = threadIdx.x & 31;
    const int l15 = lane & 15, lhi = lane >> 4;
#ifdef HAS_WMMA_F32
    v8f acc = {};
    const int mg = mtile * 16 + l15;
    const int ng = ntile * 16 + l15;
    for (int kk = 0; kk < QN; kk += 4) {
        const int kb = kk + lhi * 2;         /* f32 A/B: lane half holds K=2,3 */
        v2f a, b;
        a[0] = X[(size_t)kb * QN + mg];       a[1] = X[(size_t)(kb + 1) * QN + mg];
        b[0] = X[(size_t)kb * QN + ng];       b[1] = X[(size_t)(kb + 1) * QN + ng];
        acc = __builtin_amdgcn_wmma_f32_16x16x4_f32(false, a, false, b,
                                                    (short)0, acc, false, false);
    }
    #pragma unroll
    for (int r = 0; r < 8; ++r)
        T[(size_t)(mtile * 16 + r + lhi * 8) * QN + ng] = acc[r];
#else
    for (int r = 0; r < 8; ++r) {
        const int o = lane * 8 + r;
        const int m = mtile * 16 + (o >> 4), n = ntile * 16 + (o & 15);
        float s = 0.f;
        for (int k = 0; k < QN; ++k) s += X[(size_t)k * QN + m] * X[(size_t)k * QN + n];
        T[(size_t)m * QN + n] = s;
    }
#endif
}

/* ------------- Newton–Schulz: X' = 1.5 X - 0.5 X T  (f32 WMMA) ------------ */
__global__ __launch_bounds__(32) void ns_u_kernel(const float* __restrict__ Xall,
                                                  const float* __restrict__ Tall,
                                                  float* __restrict__ Dall) {
    const float* X = Xall + (size_t)blockIdx.z * QN * QN;
    const float* T = Tall + (size_t)blockIdx.z * QN * QN;
    float* D = Dall + (size_t)blockIdx.z * QN * QN;
    const int mtile = blockIdx.y, ntile = blockIdx.x;
    const int lane = threadIdx.x & 31;
    const int l15 = lane & 15, lhi = lane >> 4;
#ifdef HAS_WMMA_F32
    v8f acc = {};
    const int mg = mtile * 16 + l15;
    const int ng = ntile * 16 + l15;
    for (int kk = 0; kk < QN; kk += 4) {
        const int kb = kk + lhi * 2;
        v2f a, b;
        a[0] = X[(size_t)mg * QN + kb];       a[1] = X[(size_t)mg * QN + kb + 1];
        b[0] = T[(size_t)kb * QN + ng];       b[1] = T[(size_t)(kb + 1) * QN + ng];
        acc = __builtin_amdgcn_wmma_f32_16x16x4_f32(false, a, false, b,
                                                    (short)0, acc, false, false);
    }
    #pragma unroll
    for (int r = 0; r < 8; ++r) {
        const int m = mtile * 16 + r + lhi * 8;
        D[(size_t)m * QN + ng] = 1.5f * X[(size_t)m * QN + ng] - 0.5f * acc[r];
    }
#else
    for (int r = 0; r < 8; ++r) {
        const int o = lane * 8 + r;
        const int m = mtile * 16 + (o >> 4), n = ntile * 16 + (o & 15);
        float s = 0.f;
        for (int k = 0; k < QN; ++k) s += X[(size_t)m * QN + k] * T[(size_t)k * QN + n];
        D[(size_t)m * QN + n] = 1.5f * X[(size_t)m * QN + n] - 0.5f * s;
    }
#endif
}

/* ------------- bulk copy: async LDS round-trip (ASYNCcnt path) ------------ */
__global__ __launch_bounds__(256) void copy_kernel(const float4* __restrict__ src,
                                                   float4* __restrict__ dst, int n) {
    const int i = blockIdx.x * 256 + threadIdx.x;
#ifdef HAS_ASYNC_LDS
    __shared__ float4 buf[256];
    if (i < n) {
        __builtin_amdgcn_global_load_async_to_lds_b128(
            GPTR128(src + i), LPTR128(&buf[threadIdx.x]), 0, 0);
        __builtin_amdgcn_s_wait_asynccnt(0);
        __builtin_amdgcn_global_store_async_from_lds_b128(
            GPTR128(dst + i), LPTR128(&buf[threadIdx.x]), 0, 0);
        __builtin_amdgcn_s_wait_asynccnt(0);
    }
#else
    if (i < n) dst[i] = src[i];
#endif
}

/* ------------- head correction: out_head = head + 0.1*P*head -------------- */
__global__ __launch_bounds__(256) void headfix_kernel(const float* __restrict__ flat,
                                                      const float* __restrict__ Xpolar,
                                                      const int* __restrict__ needed,
                                                      const int* __restrict__ idx,
                                                      float* __restrict__ out) {
    __shared__ float head[QN];
    const int b = blockIdx.x, t = threadIdx.x;
    const float* src = flat + (size_t)b * KTOT;
    head[t] = src[t];
    __syncthreads();
    float o = head[t];
    if (needed[b]) {                              /* uniform per block */
        const float* P = Xpolar + (size_t)idx[b] * QN * QN + (size_t)t * QN;
        float s = 0.f;
        for (int j = 0; j < QN; ++j) s += P[j] * head[j];
        o += 0.1f * s;                            /* CORRECTION_RATE */
    }
    out[(size_t)b * KTOT + t] = o;
}

extern "C" void kernel_launch(void* const* d_in, const int* in_sizes, int n_in,
                              void* d_out, int out_size, void* d_ws, size_t ws_size,
                              hipStream_t stream) {
    const float* flat = (const float*)d_in[0];
    const float* W1   = (const float*)d_in[1];
    const float* b1   = (const float*)d_in[2];
    const float* W2   = (const float*)d_in[3];
    const float* b2   = (const float*)d_in[4];
    const float* W3   = (const float*)d_in[5];
    const float* b3   = (const float*)d_in[6];
    const float* ops  = (const float*)d_in[7];
    float* out = (float*)d_out;
    char*  ws  = (char*)d_ws;

    float* h1     = (float*)(ws + WS_H1);
    int*   needed = (int*)  (ws + WS_NEEDED);
    int*   idx    = (int*)  (ws + WS_IDX);
    float* Xa     = (float*)(ws + WS_X0);
    float* Xb     = (float*)(ws + WS_X1);
    float* T      = (float*)(ws + WS_T);

    init_h1_kernel<<<(BATCH * N1 + 255) / 256, 256, 0, stream>>>(b1, h1);
    gemm1_kernel<<<dim3(8, KSEGS), 256, 0, stream>>>(flat, W1, h1, KTOT / KSEGS);
    mlp_kernel<<<BATCH, 64, 0, stream>>>(h1, W2, b2, W3, b3, needed, idx);
    polar_init_kernel<<<3, 256, 0, stream>>>(ops, Xa);

    float* xs = Xa; float* xd = Xb;
    for (int it = 0; it < NS_ITERS; ++it) {
        ns_t_kernel<<<dim3(16, 16, 3), 32, 0, stream>>>(xs, T);
        ns_u_kernel<<<dim3(16, 16, 3), 32, 0, stream>>>(xs, T, xd);
        float* tmp = xs; xs = xd; xd = tmp;
    }

    copy_kernel<<<(BATCH * KTOT / 4 + 255) / 256, 256, 0, stream>>>(
        (const float4*)flat, (float4*)out, BATCH * KTOT / 4);
    headfix_kernel<<<BATCH, 256, 0, stream>>>(flat, xs, needed, idx, out);
}
